// MixtureOfExperts_24240795419344
// MI455X (gfx1250) — matrix operand order repaired and checked by reading, not compiled, not run on hardware
//
#include <hip/hip_runtime.h>
#include <math.h>

// ---------------------------------------------------------------------------
// MoE layer (LN -> top-2 router -> expert FFN w/ exact GELU -> residual add)
// MI455X / gfx1250: wave32, bf16 WMMA f32-accum, dispatch-based expert GEMMs.
// 64-token tiles (4 M-tiles): each streamed weight fragment feeds 4 WMMAs;
// B fragments are double-buffered so loads overlap WMMA execution.
// ---------------------------------------------------------------------------

#define T_TOK   8192          // B*S = 4*2048
#define DIM     1024          // D
#define HID     2048          // H
#define NEXP    8
#define LN_EPS  1e-5f
#define LB_W    0.01f
#define TM      64            // tokens per block in expert FFN

typedef unsigned short bf16_t;   // bf16 storage (bit pattern only)
typedef __attribute__((ext_vector_type(16))) __bf16 v16bf;
typedef __attribute__((ext_vector_type(8)))  float  v8f;

// ---- workspace layout (bytes) ----
#define W1BF_OFF  0ull
#define W2BF_OFF  (W1BF_OFF + (size_t)NEXP * HID * DIM * 2)
#define XNBF_OFF  (W2BF_OFF + (size_t)NEXP * HID * DIM * 2)
#define CNT_OFF   (XNBF_OFF + (size_t)T_TOK * DIM * 2)
#define PSUM_OFF  (CNT_OFF + 256)
#define TOK_OFF   (PSUM_OFF + 256)
#define GATE_OFF  (TOK_OFF + (size_t)NEXP * T_TOK * 4)

__device__ __forceinline__ bf16_t f2bf(float f) {
  union { float f; unsigned int u; } v; v.f = f;
  unsigned int r = v.u + 0x7fffu + ((v.u >> 16) & 1u);   // round-to-nearest-even
  return (bf16_t)(r >> 16);
}

// ---------------------------------------------------------------------------
__global__ void moe_init(int* counts, float* psum) {
  int t = threadIdx.x;
  if (t < NEXP) { counts[t] = 0; psum[t] = 0.f; }
}

// ---------------------------------------------------------------------------
// f32 -> bf16 weight conversion (w1 and w2), vectorized float4 -> 4x bf16.
__global__ __launch_bounds__(256) void conv_w(const float4* __restrict__ w1,
                                              const float4* __restrict__ w2,
                                              unsigned long long* __restrict__ o1,
                                              unsigned long long* __restrict__ o2,
                                              long n4) {
  long i = (long)blockIdx.x * blockDim.x + threadIdx.x;
  long stride = (long)gridDim.x * blockDim.x;
  for (; i < n4; i += stride) {
    float4 a = w1[i];
    float4 b = w2[i];
    unsigned long long pa = (unsigned long long)f2bf(a.x)
                          | ((unsigned long long)f2bf(a.y) << 16)
                          | ((unsigned long long)f2bf(a.z) << 32)
                          | ((unsigned long long)f2bf(a.w) << 48);
    unsigned long long pb = (unsigned long long)f2bf(b.x)
                          | ((unsigned long long)f2bf(b.y) << 16)
                          | ((unsigned long long)f2bf(b.z) << 32)
                          | ((unsigned long long)f2bf(b.w) << 48);
    o1[i] = pa;
    o2[i] = pb;
  }
}

// ---------------------------------------------------------------------------
// Fused LayerNorm + router. One block (256 thr) per token.
__global__ __launch_bounds__(256) void ln_router(
    const float* __restrict__ x, const float* __restrict__ gw,
    const float* __restrict__ gamma, const float* __restrict__ beta,
    bf16_t* __restrict__ xnbf, float* __restrict__ out,
    int* counts, float* psum, int* toklist, float* gatelist) {
  __shared__ float r1[256], r2[256];
  __shared__ float elog[NEXP];
  int tid = threadIdx.x;
  size_t tok = blockIdx.x;
  const float* xr = x + tok * DIM;

  float xs[4];
  float s = 0.f, s2 = 0.f;
#pragma unroll
  for (int j = 0; j < 4; ++j) {
    float v = xr[tid + j * 256];
    xs[j] = v; s += v; s2 += v * v;
  }
  r1[tid] = s; r2[tid] = s2;
  __syncthreads();
  for (int st = 128; st > 0; st >>= 1) {
    if (tid < st) { r1[tid] += r1[tid + st]; r2[tid] += r2[tid + st]; }
    __syncthreads();
  }
  float mu  = r1[0] * (1.f / DIM);
  float var = r2[0] * (1.f / DIM) - mu * mu;   // biased var, matches jnp.var
  float rs  = rsqrtf(var + LN_EPS);
  __syncthreads();

  float pl[NEXP];
#pragma unroll
  for (int e = 0; e < NEXP; ++e) pl[e] = 0.f;
#pragma unroll
  for (int j = 0; j < 4; ++j) {
    int idx = tid + j * 256;
    float v = (xs[j] - mu) * rs * gamma[idx] + beta[idx];
    xnbf[tok * DIM + idx] = f2bf(v);
    out[tok * DIM + idx]  = xs[j];           // residual
#pragma unroll
    for (int e = 0; e < NEXP; ++e) pl[e] += v * gw[e * DIM + idx];
  }
  for (int e = 0; e < NEXP; ++e) {
    r1[tid] = pl[e];
    __syncthreads();
    for (int st = 128; st > 0; st >>= 1) {
      if (tid < st) r1[tid] += r1[tid + st];
      __syncthreads();
    }
    if (tid == 0) elog[e] = r1[0];
    __syncthreads();
  }

  if (tid == 0) {
    float m = elog[0]; int i0 = 0;
    for (int e = 1; e < NEXP; ++e) if (elog[e] > m) { m = elog[e]; i0 = e; }
    float p[NEXP], sum = 0.f;
    for (int e = 0; e < NEXP; ++e) { p[e] = __expf(elog[e] - m); sum += p[e]; }
    float inv = 1.f / sum;
    for (int e = 0; e < NEXP; ++e) atomicAdd(&psum[e], p[e] * inv);
    int i1 = -1; float v1 = -1e30f;
    for (int e = 0; e < NEXP; ++e)
      if (e != i0 && elog[e] > v1) { v1 = elog[e]; i1 = e; }
    float g0 = 1.f / (1.f + __expf(v1 - m));   // softmax over top-2
    float g1 = 1.f - g0;
    int p0 = atomicAdd(&counts[i0], 1);
    toklist[i0 * T_TOK + p0] = (int)tok; gatelist[i0 * T_TOK + p0] = g0;
    int p1 = atomicAdd(&counts[i1], 1);
    toklist[i1 * T_TOK + p1] = (int)tok; gatelist[i1 * T_TOK + p1] = g1;
  }
}

// ---------------------------------------------------------------------------
// Fused expert FFN: y = (gelu(xn @ w1[e]^T) @ w2[e]^T) * gate, atomically
// accumulated into out. One block = 64 gathered tokens of one expert
// (4 M-tiles), 8 wave32s.
//  GEMM1: wave w owns h-cols [w*16, w*16+16) of each 128-wide H chunk.
//  GEMM2: wave w owns D-cols [w*128, w*128+128): 4x8 f32 accum tiles.
// Streamed B fragments are double-buffered: next load in flight during WMMA.
__global__ __launch_bounds__(256) void moe_ffn(
    const bf16_t* __restrict__ xnbf, const bf16_t* __restrict__ w1bf,
    const bf16_t* __restrict__ w2bf, const int* __restrict__ counts,
    const int* __restrict__ toklist, const float* __restrict__ gatelist,
    float* __restrict__ out) {
  __shared__ bf16_t Asm[TM * 1032];   // 64 token rows, padded stride (132 KB)
  __shared__ bf16_t Hsm[TM * 136];    // 64 x 128 h-chunk, padded (17 KB)
  const int e = blockIdx.y;
  const int cnt = counts[e];
  const int tbase = blockIdx.x * TM;
  if (tbase >= cnt) return;

  const int tid  = threadIdx.x;
  const int lane = tid & 31;
  const int wv   = tid >> 5;
  const int nlo  = lane & 15;
  const int kA   = (lane >> 4) << 3;   // 0 or 8  (A frag: two 8-elem K runs)
  const int kB   = (lane >> 4) << 4;   // 0 or 16 (unused; B frags also use kA)
  const int* tlist = toklist + e * T_TOK;
  const float* glist = gatelist + e * T_TOK;

  // ---- gather 64 xn rows (bf16) into LDS ----
  {
    int r = tid >> 2, cseg = (tid & 3) * 256;   // 4 threads per row
    int rho = tbase + r;
    uint4* dst = (uint4*)(Asm + r * 1032 + cseg);
    if (rho < cnt) {
      size_t tok = (size_t)tlist[rho];
      const uint4* src = (const uint4*)(xnbf + tok * DIM + cseg);
#pragma unroll
      for (int j = 0; j < 32; ++j) dst[j] = src[j];
    } else {
      uint4 z; z.x = z.y = z.z = z.w = 0u;
#pragma unroll
      for (int j = 0; j < 32; ++j) dst[j] = z;
    }
  }
  __syncthreads();

  const v8f vzero = {0.f, 0.f, 0.f, 0.f, 0.f, 0.f, 0.f, 0.f};
  v8f acc[4][8];
#pragma unroll
  for (int mt = 0; mt < 4; ++mt)
#pragma unroll
    for (int dt = 0; dt < 8; ++dt) acc[mt][dt] = vzero;

  for (int ch = 0; ch < HID / 128; ++ch) {
    // ---- GEMM1: h[64 x 16] (this wave's cols), K = 1024 over xn ----
    const bf16_t* brow1 =
        w1bf + ((size_t)e * HID + (size_t)ch * 128 + wv * 16 + nlo) * DIM + kA;
    if (ch + 1 < HID / 128)  // warm next chunk's w1 rows into cache
      __builtin_prefetch(brow1 + (size_t)128 * DIM, 0, 3);
    v8f hacc[4];
#pragma unroll
    for (int mt = 0; mt < 4; ++mt) hacc[mt] = vzero;

    union bfrag { v16bf v; uint4 q[2]; };
    bfrag Bm[2];
    Bm[0].q[0] = *(const uint4*)brow1;
    Bm[0].q[1] = *(const uint4*)(brow1 + 16);
#pragma unroll 4
    for (int kk = 0; kk < 32; ++kk) {
      if (kk + 1 < 32) {                       // double-buffer next B frag
        const bf16_t* br = brow1 + (kk + 1) * 32;
        Bm[(kk + 1) & 1].q[0] = *(const uint4*)br;
        Bm[(kk + 1) & 1].q[1] = *(const uint4*)(br + 16);
      }
#pragma unroll
      for (int mt = 0; mt < 4; ++mt) {
        union { v16bf v; uint4 q[2]; } A;
        const bf16_t* ar = Asm + (mt * 16 + nlo) * 1032 + kk * 32 + kA;
        A.q[0] = *(const uint4*)ar;
        A.q[1] = *(const uint4*)(ar + 16);
        hacc[mt] = __builtin_amdgcn_wmma_f32_16x16x32_bf16(
            false, A.v, false, Bm[kk & 1].v, (short)0, hacc[mt], false, false);
      }
    }
    // ---- exact GELU, h -> LDS (bf16) ----
#pragma unroll
    for (int mt = 0; mt < 4; ++mt) {
#pragma unroll
      for (int r = 0; r < 8; ++r) {
        float v = hacc[mt][r];
        float g = 0.5f * v * (1.f + erff(v * 0.70710678118654752f));
        int m = mt * 16 + r + kA;            // C/D layout: M = r + 8*(lane>=16)
        Hsm[m * 136 + wv * 16 + nlo] = f2bf(g);
      }
    }
    __syncthreads();

    // ---- GEMM2: acc[mt][dt] += h_chunk @ w2[e][dcols, chunk]^T ----
    const bf16_t* b2base =
        w2bf + ((size_t)e * DIM + wv * 128 + nlo) * HID + ch * 128 + kA;
#pragma unroll
    for (int kk2 = 0; kk2 < 4; ++kk2) {
      union { v16bf v; uint4 q[2]; } A2[4];
#pragma unroll
      for (int mt = 0; mt < 4; ++mt) {
        const bf16_t* ar = Hsm + (mt * 16 + nlo) * 136 + kk2 * 32 + kA;
        A2[mt].q[0] = *(const uint4*)ar;
        A2[mt].q[1] = *(const uint4*)(ar + 16);
      }
      union bfrag2 { v16bf v; uint4 q[2]; };
      bfrag2 B2[2];
      B2[0].q[0] = *(const uint4*)(b2base + kk2 * 32);
      B2[0].q[1] = *(const uint4*)(b2base + kk2 * 32 + 16);
#pragma unroll
      for (int dt = 0; dt < 8; ++dt) {
        if (dt + 1 < 8) {                      // double-buffer next B frag
          const bf16_t* br = b2base + (size_t)(dt + 1) * 16 * HID + kk2 * 32;
          B2[(dt + 1) & 1].q[0] = *(const uint4*)br;
          B2[(dt + 1) & 1].q[1] = *(const uint4*)(br + 16);
        }
#pragma unroll
        for (int mt = 0; mt < 4; ++mt) {
          acc[mt][dt] = __builtin_amdgcn_wmma_f32_16x16x32_bf16(
              false, A2[mt].v, false, B2[dt & 1].v, (short)0, acc[mt][dt],
              false, false);
        }
      }
    }
    __syncthreads();
  }

  // ---- gate-scale + scatter-accumulate into out ----
#pragma unroll
  for (int mt = 0; mt < 4; ++mt) {
    int toks[8]; float gts[8]; bool val[8];
#pragma unroll
    for (int r = 0; r < 8; ++r) {
      int rho = tbase + mt * 16 + r + kA;
      val[r] = rho < cnt;
      if (val[r]) { toks[r] = tlist[rho]; gts[r] = glist[rho]; }
      else        { toks[r] = 0;          gts[r] = 0.f; }
    }
#pragma unroll
    for (int dt = 0; dt < 8; ++dt) {
      int d = wv * 128 + dt * 16 + nlo;
#pragma unroll
      for (int r = 0; r < 8; ++r) {
        if (val[r])
          atomicAdd(out + (size_t)toks[r] * DIM + d, acc[mt][dt][r] * gts[r]);
      }
    }
  }
}

// ---------------------------------------------------------------------------
__global__ void lb_final(const float* __restrict__ psum, float* loss_out) {
  if (threadIdx.x == 0) {
    float a[NEXP], mean = 0.f;
    for (int e = 0; e < NEXP; ++e) { a[e] = psum[e] * (1.f / T_TOK); mean += a[e]; }
    mean *= (1.f / NEXP);
    float var = 0.f;
    for (int e = 0; e < NEXP; ++e) var += (a[e] - mean) * (a[e] - mean);
    var *= (1.f / (NEXP - 1));               // torch std -> ddof=1
    float r = sqrtf(var) / (mean + 1e-6f);
    *loss_out = r * r * LB_W;
  }
}

// ---------------------------------------------------------------------------
extern "C" void kernel_launch(void* const* d_in, const int* in_sizes, int n_in,
                              void* d_out, int out_size, void* d_ws, size_t ws_size,
                              hipStream_t stream) {
  const float* x     = (const float*)d_in[0];
  const float* gw    = (const float*)d_in[1];
  const float* w1    = (const float*)d_in[2];
  const float* w2    = (const float*)d_in[3];
  const float* gamma = (const float*)d_in[4];
  const float* beta  = (const float*)d_in[5];
  float* out = (float*)d_out;

  unsigned char* ws = (unsigned char*)d_ws;
  bf16_t* w1bf = (bf16_t*)(ws + W1BF_OFF);
  bf16_t* w2bf = (bf16_t*)(ws + W2BF_OFF);
  bf16_t* xnbf = (bf16_t*)(ws + XNBF_OFF);
  int*    counts = (int*)(ws + CNT_OFF);
  float*  psum   = (float*)(ws + PSUM_OFF);
  int*    toklist  = (int*)(ws + TOK_OFF);
  float*  gatelist = (float*)(ws + GATE_OFF);

  moe_init<<<1, 64, 0, stream>>>(counts, psum);

  long n4 = (long)NEXP * HID * DIM / 4;
  conv_w<<<2048, 256, 0, stream>>>((const float4*)w1, (const float4*)w2,
                                   (unsigned long long*)w1bf,
                                   (unsigned long long*)w2bf, n4);

  ln_router<<<T_TOK, 256, 0, stream>>>(x, gw, gamma, beta, xnbf, out,
                                       counts, psum, toklist, gatelist);

  dim3 g(T_TOK / TM, NEXP);
  moe_ffn<<<g, 256, 0, stream>>>(xnbf, w1bf, w2bf, counts, toklist, gatelist, out);

  lb_final<<<1, 32, 0, stream>>>(psum, out + (size_t)T_TOK * DIM);
}
